// DotProductAttention_32727650796335
// MI455X (gfx1250) — matrix operand order repaired and checked by reading, not compiled, not run on hardware
//
#include <hip/hip_runtime.h>

// ---------------------------------------------------------------------------
// DotProductAttention for MI455X (gfx1250), wave32 + WMMA f16 16x16x32.
//   q = query@wq.T+bq ; k = key@wk.T+bk ; v = value@wv.T+bv   (fp32, -> f16 ws)
//   S[i,j] = k_i . q_j ; P = softmax_j(S) ; out[i,:] = P @ v   (flash, WMMA)
// Shapes: B=8, L=4096, F=64.  Workspace: 3 * 8*4096*64 f16 = 12 MB.
// Attention computes S^T = Q x K^T and O^T = V^T x P^T so per-row softmax
// stats live in per-lane scalars (C/D lanes map to N = k-row index).
// Double-buffered LDS staging: next chunk's global loads issue right after
// the barrier and their latency hides under the current chunk's 16 WMMAs.
// ---------------------------------------------------------------------------

#define BATCH 8
#define SEQL  4096
#define FDIM  64
#define SBUF  (64 * 72)   // halfs per staging buffer (64 rows, stride 72)

typedef __attribute__((ext_vector_type(16))) _Float16 v16h;
typedef __attribute__((ext_vector_type(8)))  float    v8f;

union Frag16 { uint4 u[2]; v16h h; };   // 16 halfs = 8 VGPRs (A/B fragment)

static __device__ __forceinline__ unsigned pkrtz(float a, float b) {
  return __builtin_bit_cast(unsigned, __builtin_amdgcn_cvt_pkrtz(a, b));
}

static __device__ __forceinline__ float swap16f(float v, int lane) {
  return __int_as_float(
      __builtin_amdgcn_ds_bpermute((lane ^ 16) << 2, __float_as_int(v)));
}

// ---------------------------------------------------------------------------
// Projection: y[r][o] = sum_f x[r][f] * w[o][f] + bias[o], f16 output.
// One block = 64 rows; x-tile and W staged in LDS; fp32 math (memory-bound).
// ---------------------------------------------------------------------------
__global__ __launch_bounds__(256) void qkv_proj64(
    const float* __restrict__ x, const float* __restrict__ w,
    const float* __restrict__ bias, _Float16* __restrict__ y) {
  __shared__ float xs[64][64];
  __shared__ float ws[64][65];   // pad -> conflict-free ws[o][f] across lanes
  __shared__ float bs[64];
  const int  t       = threadIdx.x;
  const long rowbase = (long)blockIdx.x * 64;

  {
    const int r = t >> 2, c0 = (t & 3) << 4;
    const float4* src = (const float4*)(x + (rowbase + r) * FDIM + c0);
    float4* dst = (float4*)&xs[r][c0];
#pragma unroll
    for (int i = 0; i < 4; ++i) dst[i] = src[i];
    const float* wsrc = w + r * FDIM + c0;
#pragma unroll
    for (int i = 0; i < 16; ++i) ws[r][c0 + i] = wsrc[i];
  }
  if (t < 64) bs[t] = bias[t];
  __syncthreads();

  const int o = t & 63, rg = t >> 6;
#pragma unroll 2
  for (int rr = 0; rr < 16; ++rr) {
    const int r = (rg << 4) + rr;
    float acc = bs[o];
#pragma unroll
    for (int f = 0; f < FDIM; ++f) acc = fmaf(xs[r][f], ws[o][f], acc);
    y[(rowbase + r) * FDIM + o] = (_Float16)acc;
  }
}

// ---------------------------------------------------------------------------
// Flash attention over k-rows. 8 waves/block, 16 k-rows per wave, j-chunk 64,
// ping-pong LDS buffers, two barriers per chunk.
// ---------------------------------------------------------------------------
__global__ __launch_bounds__(256) void flash_attn_kq(
    const _Float16* __restrict__ qh, const _Float16* __restrict__ kh,
    const _Float16* __restrict__ vh, float* __restrict__ out) {
  __shared__ __align__(16) unsigned short sq[2 * SBUF];   // Q chunks, row-major
  __shared__ __align__(16) unsigned short svT[2 * SBUF];  // V chunks transposed [f][j]

  const int t    = threadIdx.x;
  const int lane = t & 31;
  const int wave = t >> 5;
  const int b    = blockIdx.y;
  const int i0   = blockIdx.x * 128 + wave * 16;
  const int m16  = lane & 15;      // row/col index inside 16-tile
  const int hs   = lane >> 4;      // lane half selector

  // Persistent K^T B-fragments: B[f, i] = k[i0+i, f]; lane n holds column i=n%16,
  // K-range f = kk*32 + hs*16 + {0..15} (contiguous 32B per lane).
  Frag16 kb[2];
  {
    const _Float16* kp = kh + ((long)(b * SEQL + i0 + m16)) * FDIM;
#pragma unroll
    for (int kk = 0; kk < 2; ++kk) {
      const uint4* p = (const uint4*)(kp + kk * 32 + hs * 16);
      kb[kk].u[0] = p[0];
      kb[kk].u[1] = p[1];
    }
  }

  v8f acc[4];                       // O^T accumulators: 4 f-tiles x 16 i
#pragma unroll
  for (int ft = 0; ft < 4; ++ft)
#pragma unroll
    for (int r = 0; r < 8; ++r) acc[ft][r] = 0.0f;
  float mrow = -3.0e38f, lrow = 0.0f;
  const float L2E = 1.4426950408889634f;

  // staging assignments (256 threads, 64-row chunk)
  const int qrow = t >> 2, qseg = (t & 3) << 1;   // Q: 64 rows x 4 seg-pairs
  const int vpr  = t >> 3, vseg = t & 7;          // V: 32 row-pairs x 8 segs
  const _Float16* qbase = qh + ((long)b * SEQL) * FDIM;
  const _Float16* vbase = vh + ((long)b * SEQL) * FDIM;

  // ---- preload chunk 0 into registers ----
  uint4 rq0, rq1, ra, rb;
  {
    const uint4* qs = (const uint4*)(qbase + (long)qrow * FDIM + qseg * 8);
    rq0 = qs[0];
    rq1 = qs[1];
    ra = *(const uint4*)(vbase + (long)(2 * vpr)     * FDIM + vseg * 8);
    rb = *(const uint4*)(vbase + (long)(2 * vpr + 1) * FDIM + vseg * 8);
  }

  for (int j0 = 0; j0 < SEQL; j0 += 128) {
#pragma unroll
    for (int ph = 0; ph < 2; ++ph) {
      unsigned short* sqb = sq  + ph * SBUF;
      unsigned short* svb = svT + ph * SBUF;

      // ---- store staged registers: Q row-major, V transposed via v_perm ----
      *(uint4*)(sqb + qrow * 72 + qseg * 8)     = rq0;
      *(uint4*)(sqb + qrow * 72 + qseg * 8 + 8) = rq1;
      {
        const unsigned* ap = (const unsigned*)&ra;
        const unsigned* bp = (const unsigned*)&rb;
#pragma unroll
        for (int e = 0; e < 4; ++e) {
          const unsigned lo = __builtin_amdgcn_perm(bp[e], ap[e], 0x05040100u);
          const unsigned hi = __builtin_amdgcn_perm(bp[e], ap[e], 0x07060302u);
          const int f = vseg * 8 + 2 * e;
          *(unsigned*)(svb + f * 72 + 2 * vpr)       = lo;   // (f,   j..j+1)
          *(unsigned*)(svb + (f + 1) * 72 + 2 * vpr) = hi;   // (f+1, j..j+1)
        }
      }
      __syncthreads();

      // ---- issue next chunk's global loads; latency hides under compute ----
      const int jn = j0 + ph * 64 + 64;
      if (jn < SEQL) {
        const uint4* qs = (const uint4*)(qbase + (long)(jn + qrow) * FDIM + qseg * 8);
        rq0 = qs[0];
        rq1 = qs[1];
        ra = *(const uint4*)(vbase + (long)(jn + 2 * vpr)     * FDIM + vseg * 8);
        rb = *(const uint4*)(vbase + (long)(jn + 2 * vpr + 1) * FDIM + vseg * 8);
      }

      // ---- S^T sub-tiles: D[M=j, N=i] = sum_f q[j,f] k[i,f], 2 WMMAs each ----
      v8f sv[4];
#pragma unroll
      for (int sub = 0; sub < 4; ++sub) {
        v8f c;
#pragma unroll
        for (int r = 0; r < 8; ++r) c[r] = 0.0f;
#pragma unroll
        for (int kk = 0; kk < 2; ++kk) {
          const unsigned short* qr = sqb + (sub * 16 + m16) * 72 + kk * 32 + hs * 16;
          Frag16 qa;
          qa.u[0] = *(const uint4*)(qr);
          qa.u[1] = *(const uint4*)(qr + 8);
          c = __builtin_amdgcn_wmma_f32_16x16x32_f16(false, qa.h, false, kb[kk].h,
                                                     (short)0, c, false, false);
        }
        sv[sub] = c;
      }

      // ---- online softmax over j (column i = lane%16 per lane) ----
      float pmax = sv[0][0];
#pragma unroll
      for (int sub = 0; sub < 4; ++sub)
#pragma unroll
        for (int r = 0; r < 8; ++r) pmax = fmaxf(pmax, sv[sub][r]);
      const float cmax  = fmaxf(pmax, swap16f(pmax, lane));
      const float mnew  = fmaxf(mrow, cmax);
      const float negml = -mnew * L2E;
      const float scale = __builtin_amdgcn_exp2f(fmaf(mrow, L2E, negml));
      mrow = mnew;

      float psum = 0.0f;
#pragma unroll
      for (int sub = 0; sub < 4; ++sub)
#pragma unroll
        for (int r = 0; r < 8; ++r) {
          const float e = __builtin_amdgcn_exp2f(fmaf(sv[sub][r], L2E, negml));
          sv[sub][r] = e;
          psum += e;
        }
      lrow = lrow * scale + (psum + swap16f(psum, lane));

#pragma unroll
      for (int ft = 0; ft < 4; ++ft)
#pragma unroll
        for (int r = 0; r < 8; ++r) acc[ft][r] *= scale;

      // ---- build two P^T B-fragments (j pr*32..pr*32+31) ----
      Frag16 pf[2];
#pragma unroll
      for (int pr = 0; pr < 2; ++pr) {
        v8f pown, psnd;
#pragma unroll
        for (int r = 0; r < 8; ++r) {
          pown[r] = hs ? sv[2 * pr + 1][r] : sv[2 * pr][r];
          psnd[r] = hs ? sv[2 * pr][r] : sv[2 * pr + 1][r];
        }
        unsigned own[4], rcv[4];
#pragma unroll
        for (int r = 0; r < 4; ++r) {
          own[r] = pkrtz(pown[2 * r], pown[2 * r + 1]);
          const unsigned snd = pkrtz(psnd[2 * r], psnd[2 * r + 1]);
          rcv[r] = (unsigned)__builtin_amdgcn_ds_bpermute((lane ^ 16) << 2, (int)snd);
        }
        unsigned* pd = (unsigned*)&pf[pr];
#pragma unroll
        for (int r = 0; r < 4; ++r) {
          pd[r]     = hs ? rcv[r] : own[r];
          pd[4 + r] = hs ? own[r] : rcv[r];
        }
      }

      // ---- O^T += V^T x P^T : 4 f-tiles x 2 j-halves ----
#pragma unroll
      for (int ft = 0; ft < 4; ++ft) {
        const unsigned short* vr = svb + (ft * 16 + m16) * 72;
#pragma unroll
        for (int jj = 0; jj < 2; ++jj) {
          Frag16 va;
          va.u[0] = *(const uint4*)(vr + jj * 32 + hs * 16);
          va.u[1] = *(const uint4*)(vr + jj * 32 + hs * 16 + 8);
          acc[ft] = __builtin_amdgcn_wmma_f32_16x16x32_f16(false, va.h, false, pf[jj].h,
                                                           (short)0, acc[ft], false, false);
        }
      }
      __syncthreads();   // protects buffer ph against next round's store
    }
  }

  // ---- normalize and write out[b, i, f] (lane holds i = lane%16, f per VGPR) ----
  const float inv = 1.0f / lrow;
  float* op = out + ((long)(b * SEQL + i0 + m16)) * FDIM + hs * 8;
#pragma unroll
  for (int ft = 0; ft < 4; ++ft) {
    float4 o1 = make_float4(acc[ft][0] * inv, acc[ft][1] * inv,
                            acc[ft][2] * inv, acc[ft][3] * inv);
    float4 o2 = make_float4(acc[ft][4] * inv, acc[ft][5] * inv,
                            acc[ft][6] * inv, acc[ft][7] * inv);
    *(float4*)(op + ft * 16)     = o1;
    *(float4*)(op + ft * 16 + 4) = o2;
  }
}

// ---------------------------------------------------------------------------
extern "C" void kernel_launch(void* const* d_in, const int* in_sizes, int n_in,
                              void* d_out, int out_size, void* d_ws, size_t ws_size,
                              hipStream_t stream) {
  const float* query = (const float*)d_in[0];
  const float* key   = (const float*)d_in[1];
  const float* value = (const float*)d_in[2];
  const float* wq    = (const float*)d_in[3];
  const float* bq    = (const float*)d_in[4];
  const float* wk    = (const float*)d_in[5];
  const float* bk    = (const float*)d_in[6];
  const float* wv    = (const float*)d_in[7];
  const float* bv    = (const float*)d_in[8];

  const long n = (long)BATCH * SEQL * FDIM;      // 2,097,152 elements
  _Float16* qh = (_Float16*)d_ws;                // 4 MB each, 12 MB total
  _Float16* kh = qh + n;
  _Float16* vh = kh + n;

  const int projBlocks = (BATCH * SEQL) / 64;    // 512
  qkv_proj64<<<projBlocks, 256, 0, stream>>>(query, wq, bq, qh);
  qkv_proj64<<<projBlocks, 256, 0, stream>>>(key,   wk, bk, kh);
  qkv_proj64<<<projBlocks, 256, 0, stream>>>(value, wv, bv, vh);

  dim3 grid(SEQL / 128, BATCH);                  // 32 x 8 blocks, 8 waves each
  flash_attn_kq<<<grid, 256, 0, stream>>>(qh, kh, vh, (float*)d_out);
}
